// RelationMultiHeadAttention_86741159510308
// MI455X (gfx1250) — matrix operand order repaired and checked by reading, not compiled, not run on hardware
//
#include <hip/hip_runtime.h>
#include <cstddef>

#define HN     16
#define NSEQ   512
#define NBAT   4
#define DMODEL 1024
#define DK     64
#define DR     128
#define MTOT   (NBAT * NSEQ) /* 2048 */

typedef __attribute__((ext_vector_type(16))) __bf16 v16bf;
typedef __attribute__((ext_vector_type(8)))  __bf16 v8bf;
typedef __attribute__((ext_vector_type(4)))  __bf16 v4bf;
typedef __attribute__((ext_vector_type(8)))  float  v8f;
typedef __attribute__((ext_vector_type(4)))  int    v4i;

// ---- CDNA5 async global->LDS copy (probe via __has_builtin; sync fallback) ----
#if defined(__has_builtin)
#if __has_builtin(__builtin_amdgcn_global_load_async_to_lds_b128) && \
    __has_builtin(__builtin_amdgcn_s_wait_asynccnt)
#define USE_ASYNC_COPY 1
#endif
#endif

#ifdef USE_ASYNC_COPY
typedef __attribute__((address_space(1))) v4i g_v4i;  // global
typedef __attribute__((address_space(3))) v4i l_v4i;  // LDS
static __device__ __forceinline__ void async_copy16(void* lds_dst,
                                                    const void* gsrc) {
  __builtin_amdgcn_global_load_async_to_lds_b128((g_v4i*)gsrc, (l_v4i*)lds_dst,
                                                 0, 0);
}
#endif

static __device__ __forceinline__ v8f wmma_bf16(v16bf a, v16bf b, v8f c) {
  return __builtin_amdgcn_wmma_f32_16x16x32_bf16(false, a, false, b, (short)0, c,
                                                 false, false);
}

static __device__ __forceinline__ v16bf cat16(v8bf lo, v8bf hi) {
  return __builtin_shufflevector(lo, hi, 0, 1, 2, 3, 4, 5, 6, 7, 8, 9, 10, 11,
                                 12, 13, 14, 15);
}

static __device__ __forceinline__ v8f vzero8() {
  v8f z;
#pragma unroll
  for (int i = 0; i < 8; ++i) z[i] = 0.0f;
  return z;
}

// A fragment (16x32 bf16, M x K) gathered from row-major fp32 global memory.
// Lane l (l<16) holds row M=l, K = {0..15}; lane l>=16 holds K={16..31}.
static __device__ __forceinline__ v16bf a_frag_g(const float* __restrict__ base,
                                                 int ld, int lane) {
  const int m  = lane & 15;
  const int hb = (lane >> 4) << 4;
  const float* p = base + (size_t)m * ld + hb;
  float4 x0 = *(const float4*)(p + 0);
  float4 x1 = *(const float4*)(p + 4);
  float4 x2 = *(const float4*)(p + 8);
  float4 x3 = *(const float4*)(p + 12);
  v16bf f;
  f[0]  = (__bf16)x0.x; f[1]  = (__bf16)x0.y; f[2]  = (__bf16)x0.z; f[3]  = (__bf16)x0.w;
  f[4]  = (__bf16)x1.x; f[5]  = (__bf16)x1.y; f[6]  = (__bf16)x1.z; f[7]  = (__bf16)x1.w;
  f[8]  = (__bf16)x2.x; f[9]  = (__bf16)x2.y; f[10] = (__bf16)x2.z; f[11] = (__bf16)x2.w;
  f[12] = (__bf16)x3.x; f[13] = (__bf16)x3.y; f[14] = (__bf16)x3.z; f[15] = (__bf16)x3.w;
  return f;
}

// A fragment from a K-contiguous bf16 LDS tile (row stride ld elems, base pre-offset by kk).
static __device__ __forceinline__ v16bf a_frag_l(const __bf16* base, int ld, int lane) {
  const __bf16* p = base + (lane & 15) * ld + ((lane >> 4) << 4);
  v8bf lo = *(const v8bf*)p;
  v8bf hi = *(const v8bf*)(p + 8);
  return cat16(lo, hi);
}

// B fragment (32x16 bf16, K x N) from a transposed [n][k] bf16 LDS tile.
// Lane l: column n = l&15, K = 16*(l>>4) + i (contiguous).
static __device__ __forceinline__ v16bf b_frag_l(const __bf16* base, int ld, int lane) {
  const __bf16* p = base + (lane & 15) * ld + ((lane >> 4) << 4);
  v8bf lo = *(const v8bf*)p;
  v8bf hi = *(const v8bf*)(p + 8);
  return cat16(lo, hi);
}

// B fragment from a natural [K][n] fp32 LDS slab (stride ld floats), cvt to bf16.
static __device__ __forceinline__ v16bf b_frag_lf32(const float* base, int ld,
                                                    int lane) {
  const float* p = base + ((lane >> 4) << 4) * ld + (lane & 15);
  v16bf f;
#pragma unroll
  for (int i = 0; i < 16; ++i) f[i] = (__bf16)p[i * ld];
  return f;
}

// B fragment gathered from natural [K][n] row-major fp32 global memory (stride ld).
// Half-wave reads 16 consecutive columns -> 64B-coalesced segments.
static __device__ __forceinline__ v16bf b_frag_gs(const float* __restrict__ base,
                                                  int ld, int lane) {
  const float* p = base + (size_t)((lane >> 4) << 4) * ld + (lane & 15);
  v16bf f;
#pragma unroll
  for (int i = 0; i < 16; ++i) f[i] = (__bf16)p[(size_t)i * ld];
  return f;
}

// ---------------------------------------------------------------------------
// GEMM: out = X[M,K] @ W[K,Nc] + bias.  MODE 0: fp32 row-major out.
// MODE 1: bf16 out in head layout [B, H, n, dk] (for qh/kh/vh).
// grid (M/64, Nc/64), block 256 (8 waves; wave -> 16m x 32n quadrant).
// W k-slabs (32x64 fp32) stream into a double-buffered LDS stage via
// GLOBAL_LOAD_ASYNC_TO_LDS_B128 (ASYNCcnt), overlapping WMMA on the live slab.
// ---------------------------------------------------------------------------
template <int MODE>
__global__ __launch_bounds__(256) void gemm_bias_kernel(
    const float* __restrict__ X, const float* __restrict__ W,
    const float* __restrict__ bias, void* __restrict__ outp, int M, int K,
    int Nc) {
  __shared__ __align__(16) float wstage[2][32 * 64];  // 2 x 8 KB
  const int tid = threadIdx.x, lane = tid & 31, wave = tid >> 5;
  const int mw = wave >> 1, nw = wave & 1;
  const int m0 = blockIdx.x * 64, n0 = blockIdx.y * 64;

  auto stage = [&](int buf, int kk) {  // 16 B per thread x 2 = one 8 KB slab
#pragma unroll
    for (int j = 0; j < 2; ++j) {
      const int o = (tid + j * 256) * 4;  // float index in [32][64] slab
      const int row = o >> 6, col = o & 63;
      const float* gsrc = W + (size_t)(kk + row) * Nc + n0 + col;
      float* ldst = &wstage[buf][o];
#ifdef USE_ASYNC_COPY
      async_copy16(ldst, gsrc);
#else
      *(float4*)ldst = *(const float4*)gsrc;
#endif
    }
  };

  v8f acc0 = vzero8(), acc1 = vzero8();
  const int T = K / 32;
  stage(0, 0);
  for (int i = 0; i < T; ++i) {
#ifdef USE_ASYNC_COPY
    __builtin_amdgcn_s_wait_asynccnt(0);
#endif
    __syncthreads();  // slab i visible to all waves; slab i-1 consumers done
    if (i + 1 < T) stage((i + 1) & 1, (i + 1) * 32);
    const float* buf = wstage[i & 1];
    v16bf aF = a_frag_g(X + (size_t)(m0 + 16 * mw) * K + i * 32, K, lane);
    v16bf b0 = b_frag_lf32(buf + 32 * nw, 64, lane);
    v16bf b1 = b_frag_lf32(buf + 32 * nw + 16, 64, lane);
    acc0 = wmma_bf16(aF, b0, acc0);
    acc1 = wmma_bf16(aF, b1, acc1);
  }

  const int hi = lane >> 4, c = lane & 15;
  const int gn0 = n0 + 32 * nw + c, gn1 = gn0 + 16;
  const float bb0 = bias[gn0], bb1 = bias[gn1];
#pragma unroll
  for (int r = 0; r < 8; ++r) {
    int gm = m0 + 16 * mw + r + 8 * hi;
    float v0 = acc0[r] + bb0, v1 = acc1[r] + bb1;
    if constexpr (MODE == 0) {
      float* O = (float*)outp;
      O[(size_t)gm * Nc + gn0] = v0;
      O[(size_t)gm * Nc + gn1] = v1;
    } else {
      __bf16* O = (__bf16*)outp;
      int b = gm >> 9, ns = gm & (NSEQ - 1);
      O[(((size_t)(b * HN + (gn0 >> 6)) * NSEQ) + ns) * DK + (gn0 & 63)] = (__bf16)v0;
      O[(((size_t)(b * HN + (gn1 >> 6)) * NSEQ) + ns) * DK + (gn1 & 63)] = (__bf16)v1;
    }
  }
}

// ---------------------------------------------------------------------------
// Fused relation attention: one workgroup per (b, m). 256 threads (8 waves).
// score_rel folds Wrk into a per-(b,m) [128x16] WrkQ (so r_k streams once);
// val_rel uses (att @ r_v) @ Wrv + brv (softmax rows sum to 1).
// ---------------------------------------------------------------------------
struct __align__(16) AttnSmem {
  __bf16 qm[HN * DK];      // qh[b,:,m,:]        [h][d]
  __bf16 wrkqT[HN * DR];   // (Wrk @ qm^T)^T     [h][c]
  __bf16 attb[HN * NSEQ];  // softmax probs bf16 [h][n]
  float  brkq[HN];         // dot(brk, qm[h])
  union {
    float score[HN * NSEQ];  // pre-softmax logits [h][n]   (phases S2-S3)
    struct {                 // phases S4 (score is dead)
      __bf16 wrvT[DK * DR];  // Wrv transposed  [d][c]
      __bf16 avrb[HN * DR];  // att @ r_v       [h][c]
      float  valr[HN * DK];  // relation value  [h][d]
    } s4;
  } u;
};

__global__ __launch_bounds__(256) void attn_kernel(
    const float* __restrict__ r_k, const float* __restrict__ r_v,
    const __bf16* __restrict__ qh, const __bf16* __restrict__ kh,
    const __bf16* __restrict__ vh, const float* __restrict__ Wrk,
    const float* __restrict__ brk, const float* __restrict__ Wrv,
    const float* __restrict__ brv, float* __restrict__ val_out) {
  __shared__ AttnSmem sm;
  const int tid = threadIdx.x, lane = tid & 31, wave = tid >> 5;
  const int bm = blockIdx.x;
  const int b = bm >> 9, m = bm & (NSEQ - 1);

  // ---- S0: stage q row-block for this m ----
  for (int i = tid; i < HN * DK; i += 256) {
    int h = i >> 6, d = i & 63;
    sm.qm[i] = qh[(((size_t)(b * HN + h)) * NSEQ + m) * DK + d];
  }
  __syncthreads();
  if (tid < HN) {
    float s = 0.f;
    for (int d = 0; d < DK; ++d) s += brk[d] * (float)sm.qm[tid * DK + d];
    sm.brkq[tid] = s;
  }

  // ---- S1: WrkQ^T = (Wrk[128,64] @ qm^T[64,16])^T  (1 wmma m-tile / wave) ----
  {
    const int row0 = wave * 16;
    v8f acc = vzero8();
    for (int kk = 0; kk < DK; kk += 32) {
      v16bf aF = a_frag_g(Wrk + (size_t)row0 * DK + kk, DK, lane);
      v16bf bF = b_frag_l(sm.qm + kk, DK, lane);
      acc = wmma_bf16(aF, bF, acc);
    }
    const int h = lane & 15, hi = lane >> 4;
#pragma unroll
    for (int r = 0; r < 8; ++r)
      sm.wrkqT[h * DR + row0 + r + 8 * hi] = (__bf16)acc[r];
  }
  __syncthreads();

  // ---- S2a: score_rel[n,h] = r_k[b,m] @ WrkQ + brkq   (WMMA, A from global) ----
  {
    const float* rkbase = r_k + ((size_t)(b * NSEQ + m)) * NSEQ * DR;
    for (int nt = wave; nt < NSEQ / 16; nt += 8) {
      if (nt + 8 < NSEQ / 16)
        __builtin_prefetch(rkbase + (size_t)((nt + 8) * 16) * DR, 0, 0);
      v8f acc;
      const float bq = sm.brkq[lane & 15];
#pragma unroll
      for (int r = 0; r < 8; ++r) acc[r] = bq;
      for (int kk = 0; kk < DR; kk += 32) {
        v16bf aF = a_frag_g(rkbase + (size_t)(nt * 16) * DR + kk, DR, lane);
        v16bf bF = b_frag_l(sm.wrkqT + kk, DR, lane);
        acc = wmma_bf16(aF, bF, acc);
      }
      const int h = lane & 15, hi = lane >> 4;
#pragma unroll
      for (int r = 0; r < 8; ++r)
        sm.u.score[h * NSEQ + nt * 16 + r + 8 * hi] = acc[r];
    }
  }
  __syncthreads();

  // ---- S2b: score += scale * (qm . kh)  (per-head dot, VALU) ----
  {
    const int h = tid >> 4, l16 = tid & 15;
    const float scale = 0.125f;  // 1/sqrt(64)
    const v8bf* qr = (const v8bf*)(sm.qm + h * DK);
    for (int n = l16; n < NSEQ; n += 16) {
      const v8bf* kr =
          (const v8bf*)(kh + (((size_t)(b * HN + h)) * NSEQ + n) * DK);
      float s = 0.f;
#pragma unroll
      for (int j = 0; j < 8; ++j) {
        v8bf kv = kr[j], qv = qr[j];
#pragma unroll
        for (int e = 0; e < 8; ++e) s += (float)qv[e] * (float)kv[e];
      }
      sm.u.score[h * NSEQ + n] += s * scale;
    }
  }
  __syncthreads();

  // ---- S3: softmax over n, write bf16 probs ----
  for (int h = wave; h < HN; h += 8) {
    const float* row = sm.u.score + h * NSEQ;
    float v[16], vmax = -1e30f;
#pragma unroll
    for (int i = 0; i < 16; ++i) {
      v[i] = row[lane + 32 * i];
      vmax = fmaxf(vmax, v[i]);
    }
    for (int off = 16; off >= 1; off >>= 1)
      vmax = fmaxf(vmax, __shfl_xor(vmax, off, 32));
    float sum = 0.f;
#pragma unroll
    for (int i = 0; i < 16; ++i) {
      v[i] = __expf(v[i] - vmax);
      sum += v[i];
    }
    for (int off = 16; off >= 1; off >>= 1) sum += __shfl_xor(sum, off, 32);
    const float inv = 1.0f / sum;
#pragma unroll
    for (int i = 0; i < 16; ++i)
      sm.attb[h * NSEQ + lane + 32 * i] = (__bf16)(v[i] * inv);
  }
  __syncthreads();

  // ---- stage Wrv^T into (dead) score region; S4a: avr = att @ r_v  (WMMA) ----
  for (int i = tid; i < DK * DR; i += 256) {
    int d = i >> 7, c = i & 127;
    sm.u.s4.wrvT[i] = (__bf16)Wrv[c * DK + d];
  }
  {
    const float* rvbase = r_v + ((size_t)(b * NSEQ + m)) * NSEQ * DR;
    const int cf = wave;  // 8 column fragments cover c = 0..127
    v8f acc = vzero8();
    for (int kk = 0; kk < NSEQ; kk += 32) {
      if (kk + 32 < NSEQ)
        __builtin_prefetch(rvbase + (size_t)(kk + 32) * DR + cf * 16, 0, 0);
      v16bf aF = a_frag_l(sm.attb + kk, NSEQ, lane);
      v16bf bF = b_frag_gs(rvbase + (size_t)kk * DR + cf * 16, DR, lane);
      acc = wmma_bf16(aF, bF, acc);
    }
    const int c = lane & 15, hi = lane >> 4;
#pragma unroll
    for (int r = 0; r < 8; ++r)
      sm.u.s4.avrb[(r + 8 * hi) * DR + cf * 16 + c] = (__bf16)acc[r];
  }
  __syncthreads();

  // ---- S4b: valr = avr @ Wrv  (waves 0..3, WMMA) ----
  if (wave < 4) {
    const int cf = wave;
    v8f acc = vzero8();
    for (int kk = 0; kk < DR; kk += 32) {
      v16bf aF = a_frag_l(sm.u.s4.avrb + kk, DR, lane);
      v16bf bF = b_frag_l(sm.u.s4.wrvT + (size_t)(cf * 16) * DR + kk, DR, lane);
      acc = wmma_bf16(aF, bF, acc);
    }
    const int c = lane & 15, hi = lane >> 4;
#pragma unroll
    for (int r = 0; r < 8; ++r)
      sm.u.s4.valr[(r + 8 * hi) * DK + cf * 16 + c] = acc[r];
  }

  // ---- S4c: val1 = att @ vh  (per-head, VALU), overlapped with S4b ----
  float v1[4];
  {
    const int h = tid >> 4, d0 = (tid & 15) * 4;
    v1[0] = v1[1] = v1[2] = v1[3] = 0.f;
    const __bf16* vbase = vh + (((size_t)(b * HN + h)) * NSEQ) * DK + d0;
    for (int n0 = 0; n0 < NSEQ; n0 += 8) {
      v8bf av = *(const v8bf*)(sm.attb + h * NSEQ + n0);
#pragma unroll
      for (int j = 0; j < 8; ++j) {
        const float a = (float)av[j];
        v4bf vv = *(const v4bf*)(vbase + (size_t)(n0 + j) * DK);
#pragma unroll
        for (int e = 0; e < 4; ++e) v1[e] += a * (float)vv[e];
      }
    }
  }
  __syncthreads();
  {
    const int h = tid >> 4, d0 = (tid & 15) * 4;
    const size_t ob = (size_t)bm * DMODEL + h * DK + d0;
#pragma unroll
    for (int e = 0; e < 4; ++e)
      val_out[ob + e] = v1[e] + sm.u.s4.valr[h * DK + d0 + e] + brv[d0 + e];
  }
}

// ---------------------------------------------------------------------------
extern "C" void kernel_launch(void* const* d_in, const int* in_sizes, int n_in,
                              void* d_out, int out_size, void* d_ws,
                              size_t ws_size, hipStream_t stream) {
  (void)in_sizes; (void)n_in; (void)out_size; (void)ws_size;
  const float* q    = (const float*)d_in[0];
  const float* k    = (const float*)d_in[1];
  const float* v    = (const float*)d_in[2];
  const float* r_k  = (const float*)d_in[3];
  const float* r_v  = (const float*)d_in[4];
  const float* Wq   = (const float*)d_in[5];
  const float* bq   = (const float*)d_in[6];
  const float* Wk   = (const float*)d_in[7];
  const float* bk   = (const float*)d_in[8];
  const float* Wv   = (const float*)d_in[9];
  const float* bv   = (const float*)d_in[10];
  const float* Wrk  = (const float*)d_in[11];
  const float* brk  = (const float*)d_in[12];
  const float* Wrv  = (const float*)d_in[13];
  const float* brv  = (const float*)d_in[14];
  const float* Wo   = (const float*)d_in[15];
  const float* bo   = (const float*)d_in[16];

  // workspace: qh/kh/vh bf16 head-layout (4 MB each) + fp32 val (8 MB)
  __bf16* qh = (__bf16*)d_ws;
  __bf16* kh = (__bf16*)((char*)d_ws + (size_t)4 * 1024 * 1024);
  __bf16* vh = (__bf16*)((char*)d_ws + (size_t)8 * 1024 * 1024);
  float* val = (float*)((char*)d_ws + (size_t)12 * 1024 * 1024);

  dim3 gg(MTOT / 64, DMODEL / 64), gb(256);
  gemm_bias_kernel<1><<<gg, gb, 0, stream>>>(q, Wq, bq, qh, MTOT, DMODEL, DMODEL);
  gemm_bias_kernel<1><<<gg, gb, 0, stream>>>(k, Wk, bk, kh, MTOT, DMODEL, DMODEL);
  gemm_bias_kernel<1><<<gg, gb, 0, stream>>>(v, Wv, bv, vh, MTOT, DMODEL, DMODEL);

  attn_kernel<<<dim3(MTOT), gb, 0, stream>>>(r_k, r_v, qh, kh, vh, Wrk, brk,
                                             Wrv, brv, val);

  gemm_bias_kernel<0><<<gg, gb, 0, stream>>>(val, Wo, bo, d_out, MTOT, DMODEL,
                                             DMODEL);
}
